// MultiHeadedAttention_87797721465296
// MI455X (gfx1250) — matrix operand order repaired and checked by reading, not compiled, not run on hardware
//
#include <hip/hip_runtime.h>

// ---------------------------------------------------------------------------
// Causal multi-head attention forward for MI455X (gfx1250, wave32, WMMA).
//   B=2, S=2048, d_model=1024, H=16, DH=64
// Pipeline:
//   [transpose weights -> f16 [n][k]]  (one-off, makes B-fragments contiguous)
//   [proj QKV -> f16 head layout]      (WMMA GEMM, LDS-staged A only)
//   [streaming flash attention]        (scores^T trick, zero-shuffle softmax)
//   [output projection -> f32]         (async global->LDS A staging)
// All matrix math on v_wmma_f32_16x16x32_f16 (f16 A/B, f32 accumulate).
// ---------------------------------------------------------------------------

typedef __attribute__((ext_vector_type(16))) _Float16 v16h;
typedef __attribute__((ext_vector_type(8)))  _Float16 v8h;
typedef __attribute__((ext_vector_type(4)))  _Float16 v4h;
typedef __attribute__((ext_vector_type(8)))  float    v8f;
typedef __attribute__((ext_vector_type(4)))  float    v4f;

#define NUM_HEADS 16
#define DMODEL    1024
#define BATCH     2
#define SEQ       2048
#define HEAD_DIM  64
#define NTOK      (BATCH * SEQ)   // 4096

__device__ __forceinline__ v8f wmma_f16(v16h a, v16h b, v8f c) {
  // D = A(16x32 f16) * B(32x16 f16) + C(16x16 f32)
  return __builtin_amdgcn_wmma_f32_16x16x32_f16(false, a, false, b,
                                                (short)0, c, false, false);
}

// lane L <-> lane L^16 exchange (group-of-32 swizzle, xor_mask=0x10, and=0x1f)
__device__ __forceinline__ float swz_xor16(float x) {
  return __int_as_float(__builtin_amdgcn_ds_swizzle(__float_as_int(x), 0x401F));
}
// broadcast value from srcLane (0..31) to all lanes
__device__ __forceinline__ float lane_bcast(float x, int srcLane) {
  return __int_as_float(__builtin_amdgcn_ds_bpermute(srcLane << 2,
                                                     __float_as_int(x)));
}
__device__ __forceinline__ float fast_exp2(float x) {
  return __builtin_amdgcn_exp2f(x);
}

// Async DMA: copy 32 contiguous bytes per lane from global memory into LDS.
// (GLOBAL_LOAD_ASYNC_TO_LDS_B128, ASYNCcnt-tracked; INST_OFFSET is added to
// both the LDS and global addresses, which matches a contiguous copy.)
__device__ __forceinline__ void async_copy_32B(unsigned lds_off,
                                               const void* gaddr) {
  asm volatile(
      "global_load_async_to_lds_b128 %0, %1, off\n\t"
      "global_load_async_to_lds_b128 %0, %1, off offset:16"
      :: "v"(lds_off), "v"(gaddr) : "memory");
}
__device__ __forceinline__ void wait_async_done() {
  asm volatile("s_wait_asynccnt 0x0" ::: "memory");
}
// low 32 bits of a flat shared-memory address are the LDS byte address
__device__ __forceinline__ unsigned lds_addr_of(const void* p) {
  return (unsigned)(unsigned long long)p;
}

// Build a 16x32 f16 A-fragment from a row-major row pointer (32 contiguous
// halves per row).  Layout: lo lanes hold K{0..15}, hi lanes hold
// K{16..31}; lane's M row = lane&15.
__device__ __forceinline__ v16h make_afrag_row(const _Float16* rowptr, int lh) {
  v8h x0 = *(const v8h*)(rowptr + 16 * lh);
  v8h x1 = *(const v8h*)(rowptr + 16 * lh + 8);
  v16h a;
#pragma unroll
  for (int j = 0; j < 8; ++j) { a[j] = x0[j]; a[8 + j] = x1[j]; }
  return a;
}

// B-fragment (32x16) from fragment-major f16 matrix M[n][k] (pitch = DMODEL):
// lane col = ll, lo lanes K 0..15 / hi lanes K 16..31 -> one 32B load.
__device__ __forceinline__ v16h make_bfrag(const _Float16* M, int col, int kk,
                                           int lh) {
  return *(const v16h*)(M + (size_t)col * DMODEL + kk + 16 * lh);
}

// ---------------------------------------------------------------------------
// Kernel 0: transpose + f16-convert one 1024x1024 weight matrix per z.
//   Wt[n][k] = (f16) W[k][n].   LDS 32x33-tiled so both sides are coalesced.
// ---------------------------------------------------------------------------
__global__ __launch_bounds__(128) void mha_prep_w_kernel(
    const float* __restrict__ Wq, const float* __restrict__ Wk,
    const float* __restrict__ Wv, const float* __restrict__ Wo,
    _Float16* __restrict__ WtQ, _Float16* __restrict__ WtK,
    _Float16* __restrict__ WtV, _Float16* __restrict__ WtO) {
  const int z = blockIdx.z;
  const float* W = (z == 0) ? Wq : (z == 1) ? Wk : (z == 2) ? Wv : Wo;
  _Float16* Wt   = (z == 0) ? WtQ : (z == 1) ? WtK : (z == 2) ? WtV : WtO;

  __shared__ _Float16 tile[32][33];   // +1 pad: conflict-free column reads

  const int tid = threadIdx.x;
  const int r0  = blockIdx.y * 32;    // k-range of source tile
  const int c0  = blockIdx.x * 32;    // n-range of source tile

  // read 32x32 f32 tile, coalesced (each thread: 8 consecutive floats)
  {
    const int r = tid >> 2, cseg = (tid & 3) * 8;
    const float* src = W + (size_t)(r0 + r) * DMODEL + c0 + cseg;
#pragma unroll
    for (int i = 0; i < 8; i += 4) {
      v4f f = *(const v4f*)(src + i);
#pragma unroll
      for (int j = 0; j < 4; ++j) tile[r][cseg + i + j] = (_Float16)f[j];
    }
  }
  __syncthreads();

  // write transposed rows, coalesced 16B stores
  {
    const int c = tid >> 2, rseg = (tid & 3) * 8;
    v8h o;
#pragma unroll
    for (int j = 0; j < 8; ++j) o[j] = tile[rseg + j][c];
    *(v8h*)(Wt + (size_t)(c0 + c) * DMODEL + r0 + rseg) = o;
  }
}

// ---------------------------------------------------------------------------
// Kernel 1: fused Q/K/V projections.  z = 0:Q, 1:K, 2:V.
//   out = X[4096,1024] @ W + bias, cast f16, scattered to head layout.
//   Q/K -> [B,H,S,DH]; V -> [B,H,DH,S] (transposed for the P@V B-fragment).
// Block: 128 threads (4 waves), 64x64 tile.  A via LDS (f32->f16, coalesced);
// B direct from fragment-major Wt (32B contiguous load per lane).
// ---------------------------------------------------------------------------
__global__ __launch_bounds__(128) void mha_proj_kernel(
    const float* __restrict__ q, const float* __restrict__ k,
    const float* __restrict__ v,
    const _Float16* __restrict__ WtQ, const _Float16* __restrict__ WtK,
    const _Float16* __restrict__ WtV,
    const float* __restrict__ bq, const float* __restrict__ bk,
    const float* __restrict__ bv,
    _Float16* __restrict__ Qh, _Float16* __restrict__ Kh,
    _Float16* __restrict__ Vt) {
  const int z = blockIdx.z;
  const float* X        = (z == 0) ? q   : (z == 1) ? k   : v;
  const _Float16* Wt    = (z == 0) ? WtQ : (z == 1) ? WtK : WtV;
  const float* bias     = (z == 0) ? bq  : (z == 1) ? bk  : bv;
  _Float16* out         = (z == 0) ? Qh  : (z == 1) ? Kh  : Vt;

  __shared__ _Float16 As[64][32];     // activation tile (f16)

  const int tid  = threadIdx.x;
  const int lane = tid & 31;
  const int wave = tid >> 5;
  const int ll   = lane & 15;
  const int lh   = lane >> 4;
  const int m0   = blockIdx.y * 64;
  const int n0   = blockIdx.x * 64;

  const int arow = tid >> 1;           // A-staging row (0..63)
  const int aseg = (tid & 1) * 16;     // A-staging column segment
  const float* asrc = X + (size_t)(m0 + arow) * DMODEL + aseg;

  v8f acc[4] = {};

#pragma unroll 2
  for (int kk = 0; kk < DMODEL; kk += 32) {
    // stage A: 64x32, f32 -> f16, packed 8B LDS stores
#pragma unroll
    for (int i = 0; i < 16; i += 4) {
      v4f f = *(const v4f*)(asrc + kk + i);
      v4h hv;
#pragma unroll
      for (int j = 0; j < 4; ++j) hv[j] = (_Float16)f[j];
      *(v4h*)&As[arow][aseg + i] = hv;
    }
    if (kk + 32 < DMODEL)
      __builtin_prefetch(asrc + kk + 32, 0, 3);   // global_prefetch_b8
    __syncthreads();

    const v16h a = make_afrag_row(&As[16 * wave + ll][0], lh);
#pragma unroll
    for (int t = 0; t < 4; ++t) {
      const v16h b = make_bfrag(Wt, n0 + 16 * t + ll, kk, lh);
      acc[t] = wmma_f16(a, b, acc[t]);
    }
    __syncthreads();
  }

  // D layout: element (m,n): lane = 16*(m>=8) + n, vgpr = m&7
#pragma unroll
  for (int t = 0; t < 4; ++t) {
    const int ng = n0 + 16 * t + ll;
    const float bb = bias[ng];
    const int h  = ng >> 6;
    const int dh = ng & 63;
#pragma unroll
    for (int vv = 0; vv < 8; ++vv) {
      const int mg = m0 + 16 * wave + vv + 8 * lh;
      const int bi = mg >> 11;
      const int s  = mg & 2047;
      const float val = acc[t][vv] + bb;
      if (z < 2)
        out[(((size_t)(bi * NUM_HEADS + h)) * SEQ + s) * HEAD_DIM + dh] =
            (_Float16)val;
      else  // V transposed: [B,H,DH,S]
        out[(((size_t)(bi * NUM_HEADS + h)) * HEAD_DIM + dh) * SEQ + s] =
            (_Float16)val;
    }
  }
}

// ---------------------------------------------------------------------------
// Kernel 2: streaming causal attention, one wave per 16-query tile.
// Computes scores TRANSPOSED: S^T = K(16x64) @ Q^T(64x16) so the f32 D-tile
// (lane = query col, vgpr = key row) is register-identical to the 16x32 f16
// A-fragment layout for the P @ V WMMA -- no cross-lane shuffle to form P.
// Online softmax in log2 domain.
// ---------------------------------------------------------------------------
__global__ __launch_bounds__(32) void mha_attn_kernel(
    const _Float16* __restrict__ Qh, const _Float16* __restrict__ Kh,
    const _Float16* __restrict__ Vt, _Float16* __restrict__ Ctx) {
  const int lane = threadIdx.x & 31;
  const int ll   = lane & 15;
  const int lh   = lane >> 4;
  const int qb   = blockIdx.x * 16;
  const int h    = blockIdx.y;
  const int b    = blockIdx.z;

  const size_t head = (size_t)(b * NUM_HEADS + h);
  const _Float16* Q = Qh + head * SEQ * HEAD_DIM;   // [S,DH]
  const _Float16* K = Kh + head * SEQ * HEAD_DIM;   // [S,DH]
  const _Float16* V = Vt + head * HEAD_DIM * SEQ;   // [DH,S]

  // Q^T B-fragments (K-dim = dh): lane col = ll, lo/hi lanes K 0..15/16..31
  const _Float16* qp = Q + (size_t)(qb + ll) * HEAD_DIM;
  const v16h qfr0 = *(const v16h*)(qp + 16 * lh);        // dh 0..31
  const v16h qfr1 = *(const v16h*)(qp + 32 + 16 * lh);   // dh 32..63

  v8f acc[4] = {};
  float m_run = -1e30f, l_run = 0.0f;
  const int qglob = qb + ll;
  const float sclog = 0.125f * 1.4426950408889634f;  // 1/sqrt(64) * log2(e)

  const int nblk = (qb >> 5) + 1;   // 32-key blocks under causal mask
  for (int blk = 0; blk < nblk; ++blk) {
    const int kb = blk * 32;

    v8f d0 = {}, d1 = {};
    {
      const _Float16* kp0 = K + (size_t)(kb + ll) * HEAD_DIM;
      const _Float16* kp1 = K + (size_t)(kb + 16 + ll) * HEAD_DIM;
      d0 = wmma_f16(make_afrag_row(kp0,      lh), qfr0, d0);
      d0 = wmma_f16(make_afrag_row(kp0 + 32, lh), qfr1, d0);
      d1 = wmma_f16(make_afrag_row(kp1,      lh), qfr0, d1);
      d1 = wmma_f16(make_afrag_row(kp1 + 32, lh), qfr1, d1);
    }

    float s0[8], s1[8];
    float mloc = -1e30f;
#pragma unroll
    for (int vv = 0; vv < 8; ++vv) {
      const int kg0 = kb + vv + 8 * lh;
      const int kg1 = kg0 + 16;
      s0[vv] = d0[vv] * sclog + ((kg0 > qglob) ? -1e9f : 0.0f);
      s1[vv] = d1[vv] * sclog + ((kg1 > qglob) ? -1e9f : 0.0f);
      mloc = fmaxf(mloc, fmaxf(s0[vv], s1[vv]));
    }
    mloc = fmaxf(mloc, swz_xor16(mloc));
    const float mnew  = fmaxf(m_run, mloc);
    const float alpha = fast_exp2(m_run - mnew);

    v16h pa;
    float psum = 0.0f;
#pragma unroll
    for (int vv = 0; vv < 8; ++vv) {
      const float e0 = fast_exp2(s0[vv] - mnew);
      const float e1 = fast_exp2(s1[vv] - mnew);
      psum += e0 + e1;
      pa[vv]     = (_Float16)e0;
      pa[8 + vv] = (_Float16)e1;
    }
    psum += swz_xor16(psum);
    l_run = l_run * alpha + psum;
    m_run = mnew;

#pragma unroll
    for (int vv = 0; vv < 8; ++vv) {
      const float av = lane_bcast(alpha, vv + 8 * lh);
      acc[0][vv] *= av; acc[1][vv] *= av;
      acc[2][vv] *= av; acc[3][vv] *= av;
    }

#pragma unroll
    for (int t = 0; t < 4; ++t) {
      const _Float16* vp = V + (size_t)(16 * t + ll) * SEQ + kb + 8 * lh;
      const v8h v0 = *(const v8h*)vp;
      const v8h v1 = *(const v8h*)(vp + 16);
      v16h vb;
#pragma unroll
      for (int j = 0; j < 8; ++j) { vb[j] = v0[j]; vb[8 + j] = v1[j]; }
      acc[t] = wmma_f16(pa, vb, acc[t]);
    }
  }

  const float linv = 1.0f / l_run;
#pragma unroll
  for (int vv = 0; vv < 8; ++vv) {
    const float lv = lane_bcast(linv, vv + 8 * lh);
    const int qrow = qb + vv + 8 * lh;
    const size_t base = ((size_t)b * SEQ + qrow) * DMODEL + h * HEAD_DIM;
#pragma unroll
    for (int t = 0; t < 4; ++t)
      Ctx[base + 16 * t + ll] = (_Float16)(acc[t][vv] * lv);
  }
}

// ---------------------------------------------------------------------------
// Kernel 3: output projection  out = Ctx[4096,1024](f16) @ Wo + bo  (f32 out)
// A staged into LDS via GLOBAL_LOAD_ASYNC_TO_LDS_B128 (ASYNCcnt-tracked DMA,
// no VGPR round-trip); B direct from fragment-major WtO.
// ---------------------------------------------------------------------------
__global__ __launch_bounds__(128) void mha_outproj_kernel(
    const _Float16* __restrict__ Ctx, const _Float16* __restrict__ WtO,
    const float* __restrict__ bo, float* __restrict__ Out) {
  __shared__ _Float16 As[64][32];

  const int tid  = threadIdx.x;
  const int lane = tid & 31;
  const int wave = tid >> 5;
  const int ll   = lane & 15;
  const int lh   = lane >> 4;
  const int m0   = blockIdx.y * 64;
  const int n0   = blockIdx.x * 64;

  const int arow = tid >> 1;
  const int aseg = (tid & 1) * 16;
  const _Float16* asrc = Ctx + (size_t)(m0 + arow) * DMODEL + aseg;
  const unsigned lds_dst = lds_addr_of(&As[arow][aseg]);

  v8f acc[4] = {};

#pragma unroll 2
  for (int kk = 0; kk < DMODEL; kk += 32) {
    // async DMA: 32B per thread, global -> LDS, no VGPR staging
    async_copy_32B(lds_dst, asrc + kk);
    if (kk + 32 < DMODEL)
      __builtin_prefetch(asrc + kk + 32, 0, 3);
    wait_async_done();
    __syncthreads();

    const v16h a = make_afrag_row(&As[16 * wave + ll][0], lh);
#pragma unroll
    for (int t = 0; t < 4; ++t) {
      const v16h b = make_bfrag(WtO, n0 + 16 * t + ll, kk, lh);
      acc[t] = wmma_f16(a, b, acc[t]);
    }
    __syncthreads();
  }

#pragma unroll
  for (int t = 0; t < 4; ++t) {
    const int ng = n0 + 16 * t + ll;
    const float bb = bo[ng];
#pragma unroll
    for (int vv = 0; vv < 8; ++vv) {
      const int mg = m0 + 16 * wave + vv + 8 * lh;
      Out[(size_t)mg * DMODEL + ng] = acc[t][vv] + bb;
    }
  }
}

// ---------------------------------------------------------------------------
extern "C" void kernel_launch(void* const* d_in, const int* in_sizes, int n_in,
                              void* d_out, int out_size, void* d_ws,
                              size_t ws_size, hipStream_t stream) {
  (void)in_sizes; (void)n_in; (void)out_size; (void)ws_size;

  const float* q  = (const float*)d_in[0];
  const float* k  = (const float*)d_in[1];
  const float* v  = (const float*)d_in[2];
  // d_in[3] is the additive causal mask; applied analytically in-kernel.
  const float* Wq = (const float*)d_in[4];
  const float* bq = (const float*)d_in[5];
  const float* Wk = (const float*)d_in[6];
  const float* bk = (const float*)d_in[7];
  const float* Wv = (const float*)d_in[8];
  const float* bv = (const float*)d_in[9];
  const float* Wo = (const float*)d_in[10];
  const float* bo = (const float*)d_in[11];

  const size_t headElems = (size_t)BATCH * NUM_HEADS * SEQ * HEAD_DIM; // 4Mi
  const size_t wElems    = (size_t)DMODEL * DMODEL;                    // 1Mi
  _Float16* Qh  = (_Float16*)d_ws;        // [B,H,S,DH]   8MB
  _Float16* Kh  = Qh + headElems;         // [B,H,S,DH]   8MB
  _Float16* Vt  = Kh + headElems;         // [B,H,DH,S]   8MB
  _Float16* Ctx = Vt + headElems;         // [4096,1024]  8MB
  _Float16* WtQ = Ctx + headElems;        // [n][k] f16   2MB each
  _Float16* WtK = WtQ + wElems;
  _Float16* WtV = WtK + wElems;
  _Float16* WtO = WtV + wElems;

  mha_prep_w_kernel<<<dim3(DMODEL / 32, DMODEL / 32, 4), 128, 0, stream>>>(
      Wq, Wk, Wv, Wo, WtQ, WtK, WtV, WtO);
  mha_proj_kernel<<<dim3(DMODEL / 64, NTOK / 64, 3), 128, 0, stream>>>(
      q, k, v, WtQ, WtK, WtV, bq, bk, bv, Qh, Kh, Vt);
  mha_attn_kernel<<<dim3(SEQ / 16, NUM_HEADS, BATCH), 32, 0, stream>>>(
      Qh, Kh, Vt, Ctx);
  mha_outproj_kernel<<<dim3(DMODEL / 64, NTOK / 64), 128, 0, stream>>>(
      Ctx, WtO, bo, (float*)d_out);
}
